// RetNetBackbone_85246510891205
// MI455X (gfx1250) — matrix operand order repaired and checked by reading, not compiled, not run on hardware
//
#include <hip/hip_runtime.h>
#include <hip/hip_bf16.h>
#include <math.h>

typedef __attribute__((ext_vector_type(16))) _Float16 v16h;
typedef __attribute__((ext_vector_type(8)))  float    v8f;

#define E_     768
#define NH_    4
#define VD_    1280
#define FF_    1280
#define L_     6
#define KD_    192
#define HD_    320
#define CH_    64
#define NC_    4
#define NTOK_  256
#define NROWS_ 65536
#define DIN_   1024

// ---------------------------------------------------------------- utilities
__device__ __forceinline__ float siluf(float x) { return x / (1.f + __expf(-x)); }
__device__ __forceinline__ float decay_of(int h) { return logf(1.f - exp2f(-5.f - (float)h)); }

__device__ __forceinline__ v8f wmma_f16(v16h a, v16h b, v8f c) {
  return __builtin_amdgcn_wmma_f32_16x16x32_f16(false, a, false, b, (short)0, c, false, false);
}

// A fragment, 16x32, row-major source (element (m,k) = p[m*ld+k])
// layout: lanes0-15 rows M, K 0..15; lanes16-31 K 16..31 (matches B fragments)
template<typename T>
__device__ __forceinline__ v16h load_a_frag(const T* p, int ld) {
  int lane = threadIdx.x & 31;
  int m = lane & 15, hi = lane >> 4;
  const T* row = p + (size_t)m * ld;
  v16h a;
#pragma unroll
  for (int v = 0; v < 8; ++v) {
    int k = (hi << 4) + (v << 1);
    a[2*v]   = (_Float16)(float)row[k];
    a[2*v+1] = (_Float16)(float)row[k+1];
  }
  return a;
}

// A fragment with a per-row scale (scale uniform within a lane)
template<typename T>
__device__ __forceinline__ v16h load_a_frag_rs(const T* p, int ld, float s) {
  int lane = threadIdx.x & 31;
  int m = lane & 15, hi = lane >> 4;
  const T* row = p + (size_t)m * ld;
  v16h a;
#pragma unroll
  for (int v = 0; v < 8; ++v) {
    int k = (hi << 4) + (v << 1);
    a[2*v]   = (_Float16)((float)row[k] * s);
    a[2*v+1] = (_Float16)((float)row[k+1] * s);
  }
  return a;
}

// A fragment from "transposed" source: element (m,k) = p[k*ld + m]
template<typename T>
__device__ __forceinline__ v16h load_at_frag(const T* p, int ld) {
  int lane = threadIdx.x & 31;
  int m = lane & 15, hi = lane >> 4;
  v16h a;
#pragma unroll
  for (int v = 0; v < 8; ++v) {
    int k = (hi << 4) + (v << 1);
    a[2*v]   = (_Float16)(float)p[(size_t)k * ld + m];
    a[2*v+1] = (_Float16)(float)p[(size_t)(k+1) * ld + m];
  }
  return a;
}

// B fragment, 32x16, from W^T: element (k,n) = w[n*ldw + k]
// layout used here (contiguous k, matches A fragments): lanes0-15 cols N, K 0..15; lanes16-31 K 16..31
template<typename T>
__device__ __forceinline__ v16h load_b_frag(const T* w, int ldw) {
  int lane = threadIdx.x & 31;
  int n = lane & 15, kb = (lane >> 4) << 4;
  const T* col = w + (size_t)n * ldw + kb;
  v16h b;
#pragma unroll
  for (int v = 0; v < 8; ++v) {
    b[2*v]   = (_Float16)(float)col[2*v];
    b[2*v+1] = (_Float16)(float)col[2*v+1];
  }
  return b;
}

// B fragment from row-major B: element (k,n) = p[k*ld + n]
template<typename T>
__device__ __forceinline__ v16h load_bt_frag(const T* p, int ld) {
  int lane = threadIdx.x & 31;
  int n = lane & 15, kb = (lane >> 4) << 4;
  v16h b;
#pragma unroll
  for (int v = 0; v < 8; ++v) {
    int k = kb + 2*v;
    b[2*v]   = (_Float16)(float)p[(size_t)k * ld + n];
    b[2*v+1] = (_Float16)(float)p[(size_t)(k+1) * ld + n];
  }
  return b;
}

// row-major B fragment with per-k scale table (tab pre-offset so tab[k_local])
template<typename T>
__device__ __forceinline__ v16h load_bt_frag_ks(const T* p, int ld, const float* tab) {
  int lane = threadIdx.x & 31;
  int n = lane & 15, kb = (lane >> 4) << 4;
  v16h b;
#pragma unroll
  for (int v = 0; v < 8; ++v) {
    int k = kb + 2*v;
    b[2*v]   = (_Float16)((float)p[(size_t)k * ld + n] * tab[k]);
    b[2*v+1] = (_Float16)((float)p[(size_t)(k+1) * ld + n] * tab[k+1]);
  }
  return b;
}

// ------------------------------------------------------- weight conversion
__global__ void f32to16_kernel(const float* __restrict__ in, _Float16* __restrict__ out,
                               size_t n)
{
  size_t stride = (size_t)gridDim.x * blockDim.x;
  for (size_t i = (size_t)blockIdx.x * blockDim.x + threadIdx.x; i < n; i += stride)
    out[i] = (_Float16)in[i];
}

// ------------------------------------------------------------- generic GEMM
// C[M,N] = A[M,K] @ W[N,K]^T ; 128x64 tile per block, 128 threads (4 waves),
// each wave owns a 32x64 strip (2 A-frags reused across 4 B-frags -> 8 WMMA/K-step).
// mode 0: Cf = acc (+bias)(+extra)      (f32 out; extra may alias Cf -> residual)
// mode 1: Ch = f16(acc (+bias))
// mode 2: Ch = f16(silu(extra) * acc)
template<typename AT>
__global__ __launch_bounds__(128)
void gemm_wmma_kernel(const AT* __restrict__ A, int lda,
                      const _Float16* __restrict__ W, int ldw,
                      float* __restrict__ Cf, _Float16* __restrict__ Ch, int ldc,
                      const float* __restrict__ bias, const float* __restrict__ extra,
                      int K, int mode)
{
  int wave = threadIdx.x >> 5, lane = threadIdx.x & 31;
  int m0 = blockIdx.x * 128 + wave * 32;
  int n0 = blockIdx.y * 64;
  v8f acc[2][4] = {{{}, {}, {}, {}}, {{}, {}, {}, {}}};
  for (int kk = 0; kk < K; kk += 32) {
    v16h a0 = load_a_frag(A + (size_t)m0 * lda + kk, lda);
    v16h a1 = load_a_frag(A + (size_t)(m0 + 16) * lda + kk, lda);
#pragma unroll
    for (int t = 0; t < 4; ++t) {
      v16h b = load_b_frag(W + (size_t)(n0 + t*16) * ldw + kk, ldw);
      acc[0][t] = wmma_f16(a0, b, acc[0][t]);
      acc[1][t] = wmma_f16(a1, b, acc[1][t]);
    }
  }
  int rb = (lane >> 4) << 3;
  int cb = n0 + (lane & 15);
#pragma unroll
  for (int mm = 0; mm < 2; ++mm) {
#pragma unroll
    for (int t = 0; t < 4; ++t) {
#pragma unroll
      for (int j = 0; j < 8; ++j) {
        size_t r = (size_t)(m0 + mm*16 + rb + j);
        size_t c = (size_t)(cb + t*16);
        float v = acc[mm][t][j];
        if (bias) v += bias[c];
        size_t idx = r * ldc + c;
        if (mode == 0) {
          if (extra) v += extra[idx];
          Cf[idx] = v;
        } else if (mode == 1) {
          Ch[idx] = (_Float16)v;
        } else {
          Ch[idx] = (_Float16)(siluf(extra[idx]) * v);
        }
      }
    }
  }
}

// ------------------------------------------------------------ norm kernels
__global__ __launch_bounds__(256)
void rmsnorm_kernel(const float* __restrict__ x, const float* __restrict__ w,
                    _Float16* __restrict__ out)
{
  __shared__ float red[256];
  size_t row = blockIdx.x;
  const float* xr = x + row * E_;
  int tid = threadIdx.x;
  float v[3], s = 0.f;
#pragma unroll
  for (int i = 0; i < 3; ++i) { v[i] = xr[tid + i*256]; s += v[i]*v[i]; }
  red[tid] = s; __syncthreads();
  for (int off = 128; off > 0; off >>= 1) {
    if (tid < off) red[tid] += red[tid + off];
    __syncthreads();
  }
  float inv = rsqrtf(red[0] / (float)E_ + 1e-6f);
#pragma unroll
  for (int i = 0; i < 3; ++i) {
    int c = tid + i*256;
    out[row * E_ + c] = (_Float16)(v[i] * inv * w[c]);
  }
}

__global__ __launch_bounds__(256)
void layernorm_kernel(const float* __restrict__ x, const float* __restrict__ w,
                      const float* __restrict__ b, _Float16* __restrict__ out)
{
  __shared__ float rs[256], rq[256];
  size_t row = blockIdx.x;
  const float* xr = x + row * E_;
  int tid = threadIdx.x;
  float v[3], s = 0.f, q = 0.f;
#pragma unroll
  for (int i = 0; i < 3; ++i) { v[i] = xr[tid + i*256]; s += v[i]; q += v[i]*v[i]; }
  rs[tid] = s; rq[tid] = q; __syncthreads();
  for (int off = 128; off > 0; off >>= 1) {
    if (tid < off) { rs[tid] += rs[tid + off]; rq[tid] += rq[tid + off]; }
    __syncthreads();
  }
  float mu = rs[0] / (float)E_;
  float var = rq[0] / (float)E_ - mu * mu;
  float inv = rsqrtf(var + 1e-5f);
#pragma unroll
  for (int i = 0; i < 3; ++i) {
    int c = tid + i*256;
    out[row * E_ + c] = (_Float16)((v[i] - mu) * inv * w[c] + b[c]);
  }
}

// ----------------------------------------------- rotary-style theta shift
// in: [row=65536][NH*KD] f32 ; out: [seq][head][tok][KD] f16, optional scale
__global__ void theta_kernel(const float* __restrict__ in, _Float16* __restrict__ out,
                             float scale)
{
  const float LOG1E4 = 9.210340371976184f;
  size_t total = (size_t)NROWS_ * 384;
  size_t stride = (size_t)gridDim.x * blockDim.x;
  for (size_t p = (size_t)blockIdx.x * blockDim.x + threadIdx.x; p < total; p += stride) {
    size_t row = p / 384;
    int pj = (int)(p % 384);
    int h = pj / 96, jj = pj % 96;
    int d = jj * 2;
    float x1 = in[row * E_ + h*KD_ + d];
    float x2 = in[row * E_ + h*KD_ + d + 1];
    int t = (int)(row & 255);
    int seq = (int)(row >> 8);
    float ang = __expf(-((float)jj / 95.f) * LOG1E4);
    float sn = sinf((float)t * ang);
    float cs = cosf((float)t * ang);
    size_t o = ((size_t)(seq*NH_ + h)*NTOK_ + t)*KD_ + d;
    out[o]   = (_Float16)((x1*cs - x2*sn) * scale);
    out[o+1] = (_Float16)((x2*cs + x1*sn) * scale);
  }
}

// --------------------------------------- token assembly epilogue (encoder)
__global__ void finish_tokens_kernel(float* __restrict__ tok, const float* __restrict__ actions,
                                     const float* __restrict__ act_w, const float* __restrict__ act_b,
                                     const float* __restrict__ temb)
{
  size_t total = (size_t)NROWS_ * E_;
  size_t stride = (size_t)gridDim.x * blockDim.x;
  for (size_t p = (size_t)blockIdx.x * blockDim.x + threadIdx.x; p < total; p += stride) {
    size_t row = p / E_;
    int c = (int)(p % E_);
    int bt = (int)(row >> 8);            // b*T + t
    int t = bt & 7;
    float a0 = actions[bt*2 + 0];
    float a1 = actions[bt*2 + 1];
    tok[p] += a0 * act_w[(size_t)c*770 + 768] + a1 * act_w[(size_t)c*770 + 769]
            + act_b[c] + temb[(size_t)t*E_ + c];
  }
}

// ------------------------------------- retention: inner (within-chunk) part
// one block per (seq, head, chunk); 128 threads (4 waves)
// qk = (qr @ kr^T) * inner_mask ; iscale = max(1, |qk|.rowsum)
// ro = (qk/iscale) @ v  ; iscale stored for cross kernel
__global__ __launch_bounds__(128)
void ret_inner_kernel(const _Float16* __restrict__ q16, const _Float16* __restrict__ k16,
                      const _Float16* __restrict__ v16, float* __restrict__ ro,
                      float* __restrict__ iscale_g)
{
  __shared__ float qkf[64*64];
  __shared__ _Float16 qkh[64*64];
  __shared__ float ptab[64], rowrs[64], iscl[64];
  int blk = blockIdx.x;
  int c = blk & 3, h = (blk >> 2) & 3, sq = blk >> 4;
  float dec = decay_of(h);
  int tid = threadIdx.x;
  if (tid < 64) {
    ptab[tid] = __expf((float)tid * dec);                // decay^e
    float r = __expf(dec);
    float rs = (1.f - __expf(dec * (float)(tid+1))) / (1.f - r); // mask rowsum
    rowrs[tid] = rsqrtf(rs);
  }
  __syncthreads();
  const _Float16* qp = q16 + ((size_t)(sq*NH_ + h)*NTOK_ + c*CH_) * KD_;
  const _Float16* kp = k16 + ((size_t)(sq*NH_ + h)*NTOK_ + c*CH_) * KD_;
  const _Float16* vp = v16 + ((size_t)sq*NTOK_ + c*CH_) * VD_ + h*HD_;
  int wave = tid >> 5, lane = tid & 31;
  int m0 = wave * 16;
  v8f acc[4] = {{}, {}, {}, {}};
  for (int kk = 0; kk < KD_; kk += 32) {
    v16h a = load_a_frag(qp + (size_t)m0*KD_ + kk, KD_);
#pragma unroll
    for (int t4 = 0; t4 < 4; ++t4) {
      v16h b = load_b_frag(kp + (size_t)(t4*16)*KD_ + kk, KD_);
      acc[t4] = wmma_f16(a, b, acc[t4]);
    }
  }
  int rb = m0 + ((lane >> 4) << 3);
  int cb = lane & 15;
#pragma unroll
  for (int t4 = 0; t4 < 4; ++t4)
#pragma unroll
    for (int j = 0; j < 8; ++j) {
      int i = rb + j, jc = t4*16 + cb;
      float mv = (i >= jc) ? ptab[i - jc] * rowrs[i] : 0.f;
      qkf[i*64 + jc] = acc[t4][j] * mv;
    }
  __syncthreads();
  if (tid < 64) {
    float ss = 0.f;
    for (int j = 0; j < 64; ++j) ss += fabsf(qkf[tid*64 + j]);
    ss = fmaxf(ss, 1.f);
    iscl[tid] = ss;
    iscale_g[((size_t)(sq*NH_ + h)*NC_ + c)*CH_ + tid] = ss;
  }
  __syncthreads();
  for (int idx = tid; idx < 4096; idx += 128)
    qkh[idx] = (_Float16)(qkf[idx] / iscl[idx >> 6]);
  __syncthreads();
  v16h a0 = load_a_frag(qkh + m0*64, 64);
  v16h a1 = load_a_frag(qkh + m0*64 + 32, 64);
  size_t rowbase = (size_t)sq*NTOK_ + c*CH_;
  for (int t20 = 0; t20 < 20; ++t20) {
    v8f o = {};
    v16h b0 = load_bt_frag(vp + t20*16, VD_);
    o = wmma_f16(a0, b0, o);
    v16h b1 = load_bt_frag(vp + (size_t)32*VD_ + t20*16, VD_);
    o = wmma_f16(a1, b1, o);
#pragma unroll
    for (int j = 0; j < 8; ++j) {
      int i = rb + j;
      ro[(rowbase + i)*VD_ + h*HD_ + t20*16 + cb] = o[j];
    }
  }
}

// --------------------------- retention: sequential kv scan + cross output
// one block per (seq, head); 256 threads (8 waves); 192x320 f32 state in LDS
// (240 KB state -- only possible on CDNA5's 320 KB-per-WGP LDS)
__global__ __launch_bounds__(256)
void ret_cross_kernel(const _Float16* __restrict__ q16, const _Float16* __restrict__ k16,
                      const _Float16* __restrict__ v16, float* __restrict__ ro,
                      const float* __restrict__ iscale_g)
{
  extern __shared__ float smem[];
  float* state = smem;             // 192*320 = 61440
  float* vidt  = smem + 61440;     // 64
  float* qidt  = vidt + 64;        // 64
  float* csc   = qidt + 64;        // 4 (1 used)
  float* cols  = csc + 4;          // 320
  int blk = blockIdx.x;
  int h = blk & 3, sq = blk >> 2;
  int tid = threadIdx.x, wave = tid >> 5, lane = tid & 31;
  float dec = decay_of(h);
  float cdecay = __expf(dec * 64.f);
  for (int i = tid; i < 61440; i += 256) state[i] = 0.f;
  if (tid < 64) {
    float r = __expf(dec);
    float rs = (1.f - __expf(dec * (float)(tid+1))) / (1.f - r);
    float last = (1.f - __expf(dec * 64.f)) / (1.f - r);
    qidt[tid] = __expf(dec * (float)(tid+1)) * last * rsqrtf(rs);
    vidt[tid] = __expf((float)(63 - tid) * dec) / last;
  }
  if (tid == 0) csc[0] = 1.f;
  __syncthreads();
  const _Float16* qbase = q16 + (size_t)(sq*NH_ + h)*NTOK_*KD_;
  const _Float16* kbase = k16 + (size_t)(sq*NH_ + h)*NTOK_*KD_;
  for (int c = 0; c < NC_; ++c) {
    const _Float16* qp = qbase + (size_t)c*CH_*KD_;
    const _Float16* kp = kbase + (size_t)c*CH_*KD_;
    const _Float16* vp = v16 + ((size_t)sq*NTOK_ + c*CH_)*VD_ + h*HD_;
    if (c > 0) {
      // ---- phase A: ro = ro*(isc/all) + ((qr*qid) @ state)/all
      int mi = wave & 3;
      int nb = (wave >> 2) * 10;
      int m0 = mi * 16;
      float cs = csc[0];
      float qs = qidt[m0 + (lane & 15)];
      v16h afr[6];
#pragma unroll
      for (int k6 = 0; k6 < 6; ++k6)
        afr[k6] = load_a_frag_rs(qp + (size_t)m0*KD_ + k6*32, KD_, qs);
      int rb = m0 + ((lane >> 4) << 3);
      int cb = lane & 15;
      size_t rowbase = (size_t)sq*NTOK_ + c*CH_;
      const float* iscp = iscale_g + ((size_t)(sq*NH_ + h)*NC_ + c)*CH_;
      for (int ni = nb; ni < nb + 10; ++ni) {
        v8f o = {};
#pragma unroll
        for (int k6 = 0; k6 < 6; ++k6) {
          v16h b = load_bt_frag(state + (size_t)(k6*32)*HD_ + ni*16, HD_);
          o = wmma_f16(afr[k6], b, o);
        }
#pragma unroll
        for (int j = 0; j < 8; ++j) {
          int i = rb + j;
          float isc = iscp[i];
          float all = fmaxf(isc, cs);
          size_t idx = (rowbase + i)*VD_ + h*HD_ + ni*16 + cb;
          ro[idx] = ro[idx] * (isc / all) + o[j] / all;
        }
      }
    }
    __syncthreads();
    // ---- phase B: state = state*cdecay + kr^T @ (v * vid)
    for (int tt = wave; tt < 240; tt += 8) {
      int mi = tt / 20, ni = tt % 20;
      int m0 = mi * 16;
      v8f d = {};
#pragma unroll
      for (int kk = 0; kk < 64; kk += 32) {
        v16h a = load_at_frag(kp + (size_t)kk*KD_ + m0, KD_);
        v16h b = load_bt_frag_ks(vp + (size_t)kk*VD_ + ni*16, VD_, vidt + kk);
        d = wmma_f16(a, b, d);
      }
      int rb = m0 + ((lane >> 4) << 3);
      int cb = ni*16 + (lane & 15);
#pragma unroll
      for (int j = 0; j < 8; ++j) {
        int m = rb + j;
        state[m*HD_ + cb] = state[m*HD_ + cb] * cdecay + d[j];
      }
    }
    __syncthreads();
    // ---- new cross scale = max(1, max_col sum_k |state|)
    for (int col = tid; col < HD_; col += 256) {
      float ssum = 0.f;
      for (int m = 0; m < KD_; ++m) ssum += fabsf(state[m*HD_ + col]);
      cols[col] = ssum;
    }
    __syncthreads();
    if (tid == 0) {
      float mx = 1.f;
      for (int col = 0; col < HD_; ++col) mx = fmaxf(mx, cols[col]);
      csc[0] = mx;
    }
    __syncthreads();
  }
}

// ------------------------- per-(token,head) RMS group norm + silu(g) gate
__global__ __launch_bounds__(128)
void gnorm_gate_kernel(const float* __restrict__ ro, _Float16* __restrict__ g16)
{
  __shared__ float red[4][128];
  size_t row = blockIdx.x;
  int tid = threadIdx.x;
  const float* rr = ro + row * VD_;
  float part[4] = {0.f, 0.f, 0.f, 0.f};
  float vals[10];
#pragma unroll
  for (int i = 0; i < 10; ++i) {
    int idx = tid + i*128;
    float v = rr[idx];
    vals[i] = v;
    part[idx / HD_] += v * v;
  }
#pragma unroll
  for (int g = 0; g < 4; ++g) red[g][tid] = part[g];
  __syncthreads();
  for (int off = 64; off > 0; off >>= 1) {
    if (tid < off) {
#pragma unroll
      for (int g = 0; g < 4; ++g) red[g][tid] += red[g][tid + off];
    }
    __syncthreads();
  }
  float inv[4];
#pragma unroll
  for (int g = 0; g < 4; ++g) inv[g] = rsqrtf(red[g][0] / (float)HD_ + 1e-6f);
#pragma unroll
  for (int i = 0; i < 10; ++i) {
    int idx = tid + i*128;
    size_t gi = row * VD_ + idx;
    float gv = (float)g16[gi];
    g16[gi] = (_Float16)(siluf(gv) * vals[i] * inv[idx / HD_]);
  }
}

// ----------------------------------------------------------------- driver
extern "C" void kernel_launch(void* const* d_in, const int* in_sizes, int n_in,
                              void* d_out, int out_size, void* d_ws, size_t ws_size,
                              hipStream_t stream)
{
  (void)in_sizes; (void)n_in; (void)out_size; (void)ws_size;
  const float* z       = (const float*)d_in[0];
  const float* actions = (const float*)d_in[1];
  const float* enc_w   = (const float*)d_in[2];
  const float* enc_b   = (const float*)d_in[3];
  const float* act_w   = (const float*)d_in[4];
  const float* act_b   = (const float*)d_in[5];
  const float* temb    = (const float*)d_in[6];
  const float* wq      = (const float*)d_in[7];
  const float* wk      = (const float*)d_in[8];
  const float* wv      = (const float*)d_in[9];
  const float* wg      = (const float*)d_in[10];
  const float* wo      = (const float*)d_in[11];
  const float* rn_w    = (const float*)d_in[12];
  const float* fc1_w   = (const float*)d_in[13];
  const float* gate_w  = (const float*)d_in[14];
  const float* fc2_w   = (const float*)d_in[15];
  const float* fn_w    = (const float*)d_in[16];
  const float* ln_w    = (const float*)d_in[17];
  const float* ln_b    = (const float*)d_in[18];
  const float* dec_w   = (const float*)d_in[19];
  const float* dec_b   = (const float*)d_in[20];

  char* ws = (char*)d_ws;
  float*    tok  = (float*)(ws);                         // 65536*768  f32
  float*    s32  = (float*)(ws + 201326592ull);          // 65536*1280 f32 scratch
  _Float16* h16  = (_Float16*)(ws + 536870912ull);       // 65536*768  f16
  _Float16* q16  = (_Float16*)(ws + 637534208ull);       // 65536*768  f16 (permuted)
  _Float16* k16  = (_Float16*)(ws + 738197504ull);       // 65536*768  f16 (permuted)
  _Float16* v16  = (_Float16*)(ws + 838860800ull);       // 65536*1280 f16
  _Float16* g16  = (_Float16*)(ws + 1006632960ull);      // 65536*1280 f16
  float*    iscl = (float*)(ws + 1174405120ull);         // 256*4*4*64 f32
  _Float16* w16  = (_Float16*)(ws + 1175453696ull);      // all weights as f16 (~85 MB)

  size_t o = 0;
  _Float16* wq16  = w16 + o; o += (size_t)L_*E_*E_;
  _Float16* wk16  = w16 + o; o += (size_t)L_*E_*E_;
  _Float16* wv16  = w16 + o; o += (size_t)L_*VD_*E_;
  _Float16* wg16  = w16 + o; o += (size_t)L_*VD_*E_;
  _Float16* wo16  = w16 + o; o += (size_t)L_*E_*VD_;
  _Float16* fc116 = w16 + o; o += (size_t)L_*FF_*E_;
  _Float16* gt16  = w16 + o; o += (size_t)L_*FF_*E_;
  _Float16* fc216 = w16 + o; o += (size_t)L_*E_*FF_;
  _Float16* enc16 = w16 + o; o += (size_t)E_*DIN_;
  _Float16* act16 = w16 + o; o += (size_t)E_*770;
  _Float16* dec16 = w16 + o; o += (size_t)DIN_*E_;

  const float kscale = 0.07216878364870323f;             // KD^-0.5
  const size_t R1_LDS = (size_t)(61440 + 64 + 64 + 4 + 320) * sizeof(float);
  dim3 b128(128), b256(256);

  // ---- one-time (per launch) weight downconversion: f32 -> f16 (L2-resident)
  f32to16_kernel<<<2048, b256, 0, stream>>>(wq,     wq16,  (size_t)L_*E_*E_);
  f32to16_kernel<<<2048, b256, 0, stream>>>(wk,     wk16,  (size_t)L_*E_*E_);
  f32to16_kernel<<<2048, b256, 0, stream>>>(wv,     wv16,  (size_t)L_*VD_*E_);
  f32to16_kernel<<<2048, b256, 0, stream>>>(wg,     wg16,  (size_t)L_*VD_*E_);
  f32to16_kernel<<<2048, b256, 0, stream>>>(wo,     wo16,  (size_t)L_*E_*VD_);
  f32to16_kernel<<<2048, b256, 0, stream>>>(fc1_w,  fc116, (size_t)L_*FF_*E_);
  f32to16_kernel<<<2048, b256, 0, stream>>>(gate_w, gt16,  (size_t)L_*FF_*E_);
  f32to16_kernel<<<2048, b256, 0, stream>>>(fc2_w,  fc216, (size_t)L_*E_*FF_);
  f32to16_kernel<<<2048, b256, 0, stream>>>(enc_w,  enc16, (size_t)E_*DIN_);
  f32to16_kernel<<<2048, b256, 0, stream>>>(act_w,  act16, (size_t)E_*770);
  f32to16_kernel<<<2048, b256, 0, stream>>>(dec_w,  dec16, (size_t)DIN_*E_);

  // encoder: h16 = f16(z @ enc_w^T + enc_b)
  gemm_wmma_kernel<float><<<dim3(NROWS_/128, E_/64), b128, 0, stream>>>(
      z, DIN_, enc16, DIN_, nullptr, h16, E_, enc_b, nullptr, DIN_, 1);
  // tok = feat @ act_w[:, :768]^T  (action cols + bias + temb added after)
  gemm_wmma_kernel<_Float16><<<dim3(NROWS_/128, E_/64), b128, 0, stream>>>(
      h16, E_, act16, 770, tok, nullptr, E_, nullptr, nullptr, E_, 0);
  finish_tokens_kernel<<<2048, b256, 0, stream>>>(tok, actions, act_w, act_b, temb);

  for (int l = 0; l < L_; ++l) {
    const _Float16* wql  = wq16  + (size_t)l * E_  * E_;
    const _Float16* wkl  = wk16  + (size_t)l * E_  * E_;
    const _Float16* wvl  = wv16  + (size_t)l * VD_ * E_;
    const _Float16* wgl  = wg16  + (size_t)l * VD_ * E_;
    const _Float16* wol  = wo16  + (size_t)l * E_  * VD_;
    const _Float16* fc1l = fc116 + (size_t)l * FF_ * E_;
    const _Float16* gtl  = gt16  + (size_t)l * FF_ * E_;
    const _Float16* fc2l = fc216 + (size_t)l * E_ * FF_;

    // ---- retention block
    rmsnorm_kernel<<<NROWS_, b256, 0, stream>>>(tok, rn_w + (size_t)l*E_, h16);
    gemm_wmma_kernel<_Float16><<<dim3(NROWS_/128, E_/64), b128, 0, stream>>>(
        h16, E_, wql, E_, s32, nullptr, E_, nullptr, nullptr, E_, 0);
    theta_kernel<<<4096, b256, 0, stream>>>(s32, q16, 1.0f);
    gemm_wmma_kernel<_Float16><<<dim3(NROWS_/128, E_/64), b128, 0, stream>>>(
        h16, E_, wkl, E_, s32, nullptr, E_, nullptr, nullptr, E_, 0);
    theta_kernel<<<4096, b256, 0, stream>>>(s32, k16, kscale);
    gemm_wmma_kernel<_Float16><<<dim3(NROWS_/128, VD_/64), b128, 0, stream>>>(
        h16, E_, wvl, E_, nullptr, v16, VD_, nullptr, nullptr, E_, 1);
    gemm_wmma_kernel<_Float16><<<dim3(NROWS_/128, VD_/64), b128, 0, stream>>>(
        h16, E_, wgl, E_, nullptr, g16, VD_, nullptr, nullptr, E_, 1);

    ret_inner_kernel<<<4096, b128, 0, stream>>>(q16, k16, v16, s32, iscl);
    ret_cross_kernel<<<1024, b256, R1_LDS, stream>>>(q16, k16, v16, s32, iscl);
    gnorm_gate_kernel<<<NROWS_, b128, 0, stream>>>(s32, g16);
    // tok += gated @ wo^T
    gemm_wmma_kernel<_Float16><<<dim3(NROWS_/128, E_/64), b128, 0, stream>>>(
        g16, VD_, wol, VD_, tok, nullptr, E_, nullptr, tok, VD_, 0);

    // ---- FFN block
    rmsnorm_kernel<<<NROWS_, b256, 0, stream>>>(tok, fn_w + (size_t)l*E_, h16);
    gemm_wmma_kernel<_Float16><<<dim3(NROWS_/128, FF_/64), b128, 0, stream>>>(
        h16, E_, fc1l, E_, s32, nullptr, FF_, nullptr, nullptr, E_, 0);
    gemm_wmma_kernel<_Float16><<<dim3(NROWS_/128, FF_/64), b128, 0, stream>>>(
        h16, E_, gtl, E_, nullptr, v16, FF_, nullptr, s32, E_, 2);
    gemm_wmma_kernel<_Float16><<<dim3(NROWS_/128, E_/64), b128, 0, stream>>>(
        v16, FF_, fc2l, FF_, tok, nullptr, E_, nullptr, tok, FF_, 0);
  }

  // final LN + decoder + residual into d_out
  layernorm_kernel<<<NROWS_, b256, 0, stream>>>(tok, ln_w, ln_b, h16);
  gemm_wmma_kernel<_Float16><<<dim3(NROWS_/128, DIN_/64), b128, 0, stream>>>(
      h16, E_, dec16, E_, (float*)d_out, nullptr, DIN_, dec_b, z, E_, 0);
}
